// GDM_57543971832235
// MI455X (gfx1250) — compile-verified, matrix operand order checked by reading, not run on hardware
//
#include <hip/hip_runtime.h>

typedef __attribute__((ext_vector_type(16))) __bf16        v16bf;
typedef __attribute__((ext_vector_type(8)))  float         v8f;
typedef __attribute__((ext_vector_type(4)))  unsigned int  uint4v;
typedef __attribute__((ext_vector_type(4)))  float         float4v;

#define HWPIX 9216
#define IMGW  96
#define LDP   40          // LDS row stride in bf16 elems (80B): 16 rows hit 16 distinct banks

__device__ __forceinline__ unsigned short f2bf(float f) {
  unsigned int u = __float_as_uint(f);
  u += 0x7FFFu + ((u >> 16) & 1u);
  return (unsigned short)(u >> 16);
}
__device__ __forceinline__ float bf2f(unsigned short h) {
  return __uint_as_float(((unsigned int)h) << 16);
}

union FragBF { uint4v u[2]; v16bf v; };

// ---------------- weight repack: f32 OIHW -> bf16 [tap][O][I] ----------------
__global__ void wt_kernel(const float* __restrict__ w, unsigned short* __restrict__ out,
                          int O, int I, int KK) {
  int idx = blockIdx.x * 256 + threadIdx.x;
  int total = O * I * KK;
  if (idx >= total) return;
  int tap = idx / (O * I);
  int rem = idx - tap * (O * I);
  int o = rem / I;
  int i = rem - o * I;
  out[idx] = f2bf(w[((size_t)o * I + i) * KK + tap]);
}

// ---------------- solve (I - diag(d)*Wlow) A = diag(d) + Wup, d = 1+diag(w) ----------------
__global__ void solveA_kernel(const float* __restrict__ w, float* __restrict__ A) {
  __shared__ float As[64][65];
  __shared__ float dv[64];
  int c = threadIdx.x;
  dv[c] = 1.0f + w[c * 64 + c];
  __syncthreads();
  for (int i = 0; i < 64; ++i) {
    float rhs = (i == c) ? dv[i] : ((c > i) ? w[i * 64 + c] : 0.0f);
    float s = 0.0f;
    for (int j = 0; j < i; ++j) s += w[i * 64 + j] * As[j][c];
    As[i][c] = rhs + dv[i] * s;
    __syncthreads();
  }
  for (int i = 0; i < 64; ++i) A[i * 64 + c] = As[i][c];
}

// expand 64x64 A into dense 256x256 channel-mix matrix (bf16)
__global__ void mix_expand_kernel(const float* __restrict__ A, unsigned short* __restrict__ M) {
  int idx = blockIdx.x * 256 + threadIdx.x;
  int i = idx >> 8, j = idx & 255;
  float v = ((i & 3) == (j & 3)) ? A[(i >> 2) * 64 + (j >> 2)] : 0.0f;
  M[idx] = f2bf(v);
}

// ---------------- implicit-GEMM conv, Cin=256 fixed, K templated ----------------
// 64x64 tile, BK=32 == one v_wmma_f32_16x16x32_bf16 step. 128 threads = 4 waves (2x2).
// Double-buffered LDS; A (weights) streamed via global_load_async_to_lds_b128 (ASYNCcnt),
// B (pixels) staged in VGPRs for step s+1 while step s WMMAs execute.
template<int K, bool IN_F32, bool RELU, bool RESID, bool OUT_F32>
__global__ __launch_bounds__(128)
void conv_wmma_kernel(const void* __restrict__ inBase,          // [B][256][9216] f32 or bf16
                      const unsigned short* __restrict__ Wt,    // bf16 [K*K][Cout][256]
                      const float* __restrict__ scale,
                      const float* __restrict__ bias,
                      const unsigned short* __restrict__ resid, // bf16 [B][outC][9216]
                      void* __restrict__ outBase,               // bf16 or f32
                      int Cout, int outC, int coShift) {
  constexpr int TAPS = K * K;
  constexpr int S    = TAPS * 8;          // 8 K-steps of 32 over Cin=256
  constexpr int PAD  = K >> 1;

  __shared__ unsigned short Atile[2][64 * LDP];
  __shared__ unsigned short Btile[2][64 * LDP];

  const int n0  = blockIdx.x * 64;
  const int co0 = blockIdx.y * 64;
  const int b   = blockIdx.z;

  const int t     = threadIdx.x;
  const int lane  = t & 31;
  const int wave  = t >> 5;
  const int wm    = (wave >> 1) * 32;
  const int wn    = (wave & 1) * 32;
  const int halfl = lane >> 4;
  const int lrow  = lane & 15;

  // B loader mapping: thread owns k-row b_k and 16 consecutive pixels at b_n
  const int b_k   = t >> 2;
  const int b_n   = (t & 3) * 16;
  const int pbase = n0 + b_n;
  const int h0    = pbase / IMGW;          // only division in the kernel
  const int w0    = pbase - h0 * IMGW;

  // LDS byte offsets of the two A buffers (low 32 bits of generic addr == LDS offset)
  const unsigned aBase[2] = {
    (unsigned)(unsigned long long)(const void*)&Atile[0][0],
    (unsigned)(unsigned long long)(const void*)&Atile[1][0] };

  unsigned short br[16];                   // staged B pixels (bf16)

  // ---- async A-tile: 256 chunks of 16B (64 rows x 4), 2 per thread ----
  auto issueA = [&](int buf, int s) {
    const int tap = s >> 3;
    const int k0  = (s & 7) << 5;
#pragma unroll
    for (int it = 0; it < 2; ++it) {
      const int c   = t + it * 128;
      const int row = c >> 2;
      const int cc  = c & 3;
      const unsigned lds  = aBase[buf] + (unsigned)(row * (LDP * 2) + cc * 16);
      const unsigned goff = (unsigned)((((unsigned)tap * (unsigned)Cout + (unsigned)(co0 + row)) * 256u
                                        + (unsigned)(k0 + cc * 8)) * 2u);
      asm volatile("global_load_async_to_lds_b128 %0, %1, %2"
                   :: "v"(lds), "v"(goff), "s"(Wt) : "memory");
    }
  };

  auto loadB = [&](int s) {
    const int tap = s >> 3;
    const int k0  = (s & 7) << 5;
    const int ci  = k0 + b_k;
    if (IN_F32) {
      const float* src = (const float*)inBase + ((size_t)b * 256 + ci) * HWPIX;
      if (K == 1) {
        const float4v* v4 = (const float4v*)(src + pbase);
#pragma unroll
        for (int g = 0; g < 4; ++g) {
          float4v f = v4[g];
#pragma unroll
          for (int e = 0; e < 4; ++e) br[g * 4 + e] = f2bf(f[e]);
        }
      } else {
        const int dy = tap / K - PAD, dx = tap % K - PAD;
        const int poff = dy * IMGW + dx;
        int hc = h0, wc = w0;
#pragma unroll
        for (int i = 0; i < 16; ++i) {
          bool ok = ((unsigned)(hc + dy) < (unsigned)IMGW) && ((unsigned)(wc + dx) < (unsigned)IMGW);
          br[i] = ok ? f2bf(src[pbase + i + poff]) : (unsigned short)0;
          if (++wc == IMGW) { wc = 0; ++hc; }
        }
      }
    } else {
      const unsigned short* src = (const unsigned short*)inBase + ((size_t)b * 256 + ci) * HWPIX;
      if (K == 1) {
        const uint4v* v4 = (const uint4v*)(src + pbase);
#pragma unroll
        for (int g = 0; g < 2; ++g) {
          uint4v u = v4[g];
#pragma unroll
          for (int e = 0; e < 4; ++e) {
            br[g * 8 + e * 2 + 0] = (unsigned short)(u[e] & 0xFFFFu);
            br[g * 8 + e * 2 + 1] = (unsigned short)(u[e] >> 16);
          }
        }
      } else {
        const int dy = tap / K - PAD, dx = tap % K - PAD;
        const int poff = dy * IMGW + dx;
        int hc = h0, wc = w0;
#pragma unroll
        for (int i = 0; i < 16; ++i) {
          bool ok = ((unsigned)(hc + dy) < (unsigned)IMGW) && ((unsigned)(wc + dx) < (unsigned)IMGW);
          br[i] = ok ? src[pbase + i + poff] : (unsigned short)0;
          if (++wc == IMGW) { wc = 0; ++hc; }
        }
      }
    }
  };

  auto storeB = [&](int buf) {
#pragma unroll
    for (int i = 0; i < 16; ++i)
      Btile[buf][(b_n + i) * LDP + b_k] = br[i];
  };

  v8f acc[2][2] = {};

  // ---- software pipeline: prologue ----
  issueA(0, 0);
  loadB(0);
  storeB(0);

  for (int s = 0; s < S; ++s) {
    const int cur = s & 1;
    const int nxt = cur ^ 1;
    const bool more = (s + 1 < S);

    if (more) { issueA(nxt, s + 1); loadB(s + 1); }

    // current A tile done (in-order async completion); next tile stays in flight
    if (more) asm volatile("s_wait_asynccnt 0x2" ::: "memory");
    else      asm volatile("s_wait_asynccnt 0x0" ::: "memory");
    __syncthreads();

    // fragments per ISA 16-bit lane layout:
    // lanes 0-15 hold K 0-7 & 16-23, lanes 16-31 hold K 8-15 & 24-31
    FragBF afr[2], bfr[2];
#pragma unroll
    for (int sm = 0; sm < 2; ++sm) {
      const unsigned short* ar = &Atile[cur][(wm + sm * 16 + lrow) * LDP];
      afr[sm].u[0] = *(const uint4v*)(ar + halfl * 8);
      afr[sm].u[1] = *(const uint4v*)(ar + 16 + halfl * 8);
    }
#pragma unroll
    for (int sn = 0; sn < 2; ++sn) {
      const unsigned short* brw = &Btile[cur][(wn + sn * 16 + lrow) * LDP];
      bfr[sn].u[0] = *(const uint4v*)(brw + halfl * 8);
      bfr[sn].u[1] = *(const uint4v*)(brw + 16 + halfl * 8);
    }
#pragma unroll
    for (int sm = 0; sm < 2; ++sm)
#pragma unroll
      for (int sn = 0; sn < 2; ++sn)
        acc[sm][sn] = __builtin_amdgcn_wmma_f32_16x16x32_bf16(
            false, afr[sm].v, false, bfr[sn].v, (short)0, acc[sm][sn], false, false);

    __syncthreads();   // all waves done reading `cur` (and guards next async-A into it)
    if (more) storeB(nxt);
  }

  // ---- epilogue: BN scale/bias, ReLU, optional residual, store ----
#pragma unroll
  for (int sm = 0; sm < 2; ++sm) {
#pragma unroll
    for (int r = 0; r < 8; ++r) {
      const int coL = co0 + wm + sm * 16 + halfl * 8 + r;   // D: m = r + 8*(lane>=16)
      float sc = 1.0f, bi = 0.0f;
      if (scale) { sc = scale[coL]; bi = bias[coL]; }
#pragma unroll
      for (int sn = 0; sn < 2; ++sn) {
        const int n = n0 + wn + sn * 16 + lrow;
        float v = acc[sm][sn][r];
        v = v * sc + bi;
        if (RELU) v = fmaxf(v, 0.0f);
        const size_t idx = ((size_t)b * outC + coShift + coL) * HWPIX + n;
        if (RESID) v += bf2f(resid[idx]);
        if (OUT_F32) ((float*)outBase)[idx] = v;
        else         ((unsigned short*)outBase)[idx] = f2bf(v);
      }
    }
  }
}

extern "C" void kernel_launch(void* const* d_in, const int* in_sizes, int n_in,
                              void* d_out, int out_size, void* d_ws, size_t ws_size,
                              hipStream_t stream) {
  (void)in_sizes; (void)n_in; (void)out_size; (void)ws_size;
  const float* x1     = (const float*)d_in[0];
  const float* x2     = (const float*)d_in[1];
  const float* weight = (const float*)d_in[2];
  const float* r1_w = (const float*)d_in[3],  *r1_s = (const float*)d_in[4],  *r1_b = (const float*)d_in[5];
  const float* r2_w = (const float*)d_in[6],  *r2_s = (const float*)d_in[7],  *r2_b = (const float*)d_in[8];
  const float* ck_w = (const float*)d_in[9],  *ck_s = (const float*)d_in[10], *ck_b = (const float*)d_in[11];
  const float* c1_w = (const float*)d_in[12], *c1_s = (const float*)d_in[13], *c1_b = (const float*)d_in[14];
  const float* cf_w = (const float*)d_in[15], *cf_s = (const float*)d_in[16], *cf_b = (const float*)d_in[17];

  const size_t planeElems = (size_t)4 * 256 * 9216;
  char* p = (char*)d_ws;
  unsigned short* xcat = (unsigned short*)p; p += planeElems * 2;   // x (later reused as z)
  unsigned short* xm   = (unsigned short*)p; p += planeElems * 2;
  unsigned short* ybuf = (unsigned short*)p; p += planeElems * 2;
  unsigned short* wr1  = (unsigned short*)p; p += (size_t)128 * 256 * 2;
  unsigned short* wr2  = (unsigned short*)p; p += (size_t)128 * 256 * 2;
  unsigned short* wmix = (unsigned short*)p; p += (size_t)256 * 256 * 2;
  unsigned short* wck  = (unsigned short*)p; p += (size_t)9 * 256 * 256 * 2;
  unsigned short* wc1  = (unsigned short*)p; p += (size_t)256 * 256 * 2;
  unsigned short* wcf  = (unsigned short*)p; p += (size_t)9 * 256 * 256 * 2;
  float* Amat = (float*)p;

  wt_kernel<<<dim3((128 * 256 + 255) / 256), 256, 0, stream>>>(r1_w, wr1, 128, 256, 1);
  wt_kernel<<<dim3((128 * 256 + 255) / 256), 256, 0, stream>>>(r2_w, wr2, 128, 256, 1);
  wt_kernel<<<dim3((256 * 256 * 9 + 255) / 256), 256, 0, stream>>>(ck_w, wck, 256, 256, 9);
  wt_kernel<<<dim3((256 * 256 + 255) / 256), 256, 0, stream>>>(c1_w, wc1, 256, 256, 1);
  wt_kernel<<<dim3((256 * 256 * 9 + 255) / 256), 256, 0, stream>>>(cf_w, wcf, 256, 256, 9);
  solveA_kernel<<<1, 64, 0, stream>>>(weight, Amat);
  mix_expand_kernel<<<256, 256, 0, stream>>>(Amat, wmix);

  dim3 blk(128);
  // reduce1 / reduce2 (f32 in, bf16 out, concat via channel shift)
  conv_wmma_kernel<1, true,  true,  false, false><<<dim3(144, 2, 4), blk, 0, stream>>>(
      x1, wr1, r1_s, r1_b, nullptr, xcat, 128, 256, 0);
  conv_wmma_kernel<1, true,  true,  false, false><<<dim3(144, 2, 4), blk, 0, stream>>>(
      x2, wr2, r2_s, r2_b, nullptr, xcat, 128, 256, 128);
  // chunk mix (pure linear 1x1)
  conv_wmma_kernel<1, false, false, false, false><<<dim3(144, 4, 4), blk, 0, stream>>>(
      xcat, wmix, nullptr, nullptr, nullptr, xm, 256, 256, 0);
  // conv_chunk 3x3 + BNR + residual add of x
  conv_wmma_kernel<3, false, true,  true,  false><<<dim3(144, 4, 4), blk, 0, stream>>>(
      xm, wck, ck_s, ck_b, xcat, ybuf, 256, 256, 0);
  // conv_1 1x1 + BNR (z overwrites xcat)
  conv_wmma_kernel<1, false, true,  false, false><<<dim3(144, 4, 4), blk, 0, stream>>>(
      ybuf, wc1, c1_s, c1_b, nullptr, xcat, 256, 256, 0);
  // final conv 3x3 + BNR, f32 output
  conv_wmma_kernel<3, false, true,  false, true ><<<dim3(144, 4, 4), blk, 0, stream>>>(
      xcat, wcf, cf_s, cf_b, nullptr, d_out, 256, 256, 0);
}
